// VectorQuantizeSimple_27633819583046
// MI455X (gfx1250) — matrix operand order repaired and checked by reading, not compiled, not run on hardware
//
#include <hip/hip_runtime.h>
#include <math.h>

// ---------------------------------------------------------------------------
// VQ codebook lookup on MI455X (gfx1250, wave32, WMMA).
// scores = z_norm(32768x512) . code_norm^T(512x8192), fused row-argmax,
// gather z_q = c_norm[idx], loss = 1.25 * mean((z_q - z)^2).
// v2: K-split dual accumulators (2 independent WMMA chains per wave),
//     single barrier per N-tile with double-buffered LDS.
// ---------------------------------------------------------------------------

#define NE     8192
#define ED     512
#define ROWS   32768          // 32*1024
#define M_TILE 128            // rows per block (8 waves x 16 rows)
#define NT     (NE / 16)      // 512 column tiles of 16
#define BPAD   520            // padded LDS row stride (ushorts) -> conflict-free

typedef __attribute__((ext_vector_type(16))) __bf16          v16bf;
typedef __attribute__((ext_vector_type(8)))  float           v8f;
typedef __attribute__((ext_vector_type(8)))  unsigned short  ushort8;

union Frag { v16bf v; ushort8 h[2]; };

__device__ __forceinline__ unsigned short f32_to_bf16(float f) {
  union { float f; unsigned u; } v; v.f = f;
  unsigned r = v.u + 0x7FFFu + ((v.u >> 16) & 1u);  // round-to-nearest-even
  return (unsigned short)(r >> 16);
}

// --------------------------- row L2-normalize ------------------------------
// one block per row; 256 threads, 2 elems/thread. Writes f32 (optional) + bf16.
__global__ void __launch_bounds__(256)
vq_normalize_kernel(const float* __restrict__ src,
                    float* __restrict__ dst_f32,
                    unsigned short* __restrict__ dst_bf16) {
  __shared__ float red[256];
  const int row = blockIdx.x;
  const int tid = threadIdx.x;
  const float* s = src + (size_t)row * ED;
  const float x0 = s[tid];
  const float x1 = s[tid + 256];
  red[tid] = x0 * x0 + x1 * x1;
  __syncthreads();
  for (int off = 128; off > 0; off >>= 1) {
    if (tid < off) red[tid] += red[tid + off];
    __syncthreads();
  }
  const float inv = 1.0f / fmaxf(sqrtf(red[0]), 1e-12f);
  const float y0 = x0 * inv, y1 = x1 * inv;
  if (dst_f32) {
    dst_f32[(size_t)row * ED + tid]       = y0;
    dst_f32[(size_t)row * ED + tid + 256] = y1;
  }
  dst_bf16[(size_t)row * ED + tid]       = f32_to_bf16(y0);
  dst_bf16[(size_t)row * ED + tid + 256] = f32_to_bf16(y1);
}

// ------------------- fused GEMM + argmax + gather + loss -------------------
__global__ void __launch_bounds__(256)
vq_argmax_gemm_kernel(const unsigned short* __restrict__ zbf,   // [ROWS][ED] bf16, normalized
                      const unsigned short* __restrict__ cbf,   // [NE][ED]  bf16, normalized
                      const float* __restrict__ cnorm,          // [NE][ED]  f32,  normalized
                      const float* __restrict__ z,              // original z f32
                      float* __restrict__ out_zq,               // [ROWS][ED]
                      float* __restrict__ out_idx,              // [ROWS] (as f32)
                      float* __restrict__ partials) {           // [gridDim.x]
  __shared__ unsigned short Bt[2][16 * BPAD];   // double-buffered B tile
  __shared__ float bestS[M_TILE * 16];
  __shared__ int   bestI[M_TILE * 16];
  __shared__ int   rowIdx[M_TILE];
  __shared__ float lred[256];

  const int tid    = threadIdx.x;
  const int wave   = tid >> 5;
  const int lane   = tid & 31;
  const int lane16 = lane & 15;
  const int half   = lane >> 4;
  const int blockRow = blockIdx.x * M_TILE;

  // ---- A fragments (this wave's 16 rows x K=512) resident in VGPRs ----
  // ISA 7.12.2: 16-bit A 16x32: lanes0-15 row M=lane, K=[0..7],[16..23];
  //             lanes16-31 row M=lane-16, K=[8..15],[24..31].
  Frag a[16];
  {
    const unsigned short* zr =
        zbf + (size_t)(blockRow + wave * 16 + lane16) * ED;
    const int koff = half * 8;
#pragma unroll
    for (int ks = 0; ks < 16; ++ks) {
      a[ks].h[0] = *(const ushort8*)(zr + ks * 32 + koff);
      a[ks].h[1] = *(const ushort8*)(zr + ks * 32 + koff + 16);
    }
  }

  float best[8];
  int   bidx[8];
#pragma unroll
  for (int j = 0; j < 8; ++j) { best[j] = -3.402823466e38f; bidx[j] = 0; }

  // ---- stage first B tile (16 codebook rows x 512) into LDS ----
#pragma unroll
  for (int j = 0; j < 4; ++j) {
    const int i = j * 2048 + tid * 8;           // flat elem in 16x512 tile
    const ushort8 v = *(const ushort8*)(cbf + i);
    *(ushort8*)&Bt[0][(i >> 9) * BPAD + (i & 511)] = v;
  }
  __syncthreads();

  ushort8 pf[4];
  for (int t = 0; t < NT; ++t) {
    const int buf = t & 1;

    // prefetch next tile into registers (code_bf16 is L2-resident: 8 MB)
    if (t + 1 < NT) {
      const unsigned short* g = cbf + (size_t)(t + 1) * (16 * ED);
#pragma unroll
      for (int j = 0; j < 4; ++j)
        pf[j] = *(const ushort8*)(g + j * 2048 + tid * 8);
      if (t + 2 < NT)
        __builtin_prefetch(cbf + (size_t)(t + 2) * (16 * ED) + tid * 8, 0, 1);
    }

    // ---- 16x16 output tile, K split into two independent WMMA chains ----
    // B layout: lane n<16 holds col N=n, K=[0..7],[16..23]; lanes 16-31 hold
    // col N=lane-16, K=[8..15],[24..31] of each K-step (matches A).
    v8f acc0 = {};
    v8f acc1 = {};
    {
      const unsigned short* brow = &Bt[buf][lane16 * BPAD + half * 8];
#pragma unroll
      for (int ks = 0; ks < 8; ++ks) {
        Frag b0, b1;
        b0.h[0] = *(const ushort8*)(brow + ks * 32);
        b0.h[1] = *(const ushort8*)(brow + ks * 32 + 16);
        b1.h[0] = *(const ushort8*)(brow + (ks + 8) * 32);
        b1.h[1] = *(const ushort8*)(brow + (ks + 8) * 32 + 16);
        acc0 = __builtin_amdgcn_wmma_f32_16x16x32_bf16(
            false, a[ks].v, false, b0.v, (short)0, acc0, false, false);
        acc1 = __builtin_amdgcn_wmma_f32_16x16x32_bf16(
            false, a[ks + 8].v, false, b1.v, (short)0, acc1, false, false);
      }
    }

    // running argmax (strict > keeps earliest == smallest index on ties)
    const int ncol = t * 16 + lane16;
#pragma unroll
    for (int j = 0; j < 8; ++j) {
      const float s = acc0[j] + acc1[j];
      if (s > best[j]) { best[j] = s; bidx[j] = ncol; }
    }

    // store prefetched tile into the other buffer: its last readers were
    // fenced by the barrier ending iteration t-1 -> one barrier per tile.
    if (t + 1 < NT) {
      const int b2 = (t + 1) & 1;
#pragma unroll
      for (int j = 0; j < 4; ++j) {
        const int i = j * 2048 + tid * 8;
        *(ushort8*)&Bt[b2][(i >> 9) * BPAD + (i & 511)] = pf[j];
      }
    }
    __syncthreads();
  }

  // ---- cross-lane argmax: D layout: lane n=lane&15, VGPR j -> M=j+8*half ----
#pragma unroll
  for (int j = 0; j < 8; ++j) {
    const int m = wave * 16 + half * 8 + j;
    bestS[m * 16 + lane16] = best[j];
    bestI[m * 16 + lane16] = bidx[j];
  }
  __syncthreads();

  if (tid < M_TILE) {
    float bs = bestS[tid * 16];
    int   bi = bestI[tid * 16];
#pragma unroll
    for (int r = 1; r < 16; ++r) {
      const float s = bestS[tid * 16 + r];
      const int   i = bestI[tid * 16 + r];
      if (s > bs || (s == bs && i < bi)) { bs = s; bi = i; }
    }
    rowIdx[tid] = bi;
    out_idx[blockRow + tid] = (float)bi;       // argmin distance == argmax dot
  }
  __syncthreads();

  // ---- coalesced gather z_q = c_norm[idx] + loss partial ----
  float lsum = 0.0f;
  for (int r = 0; r < M_TILE; ++r) {
    const int n = rowIdx[r];
    const size_t row = (size_t)(blockRow + r);
    const float* cq = cnorm + (size_t)n * ED;
    const float* zp = z + row * ED;
    float* op = out_zq + row * ED;
#pragma unroll
    for (int k = 0; k < 2; ++k) {
      const int i = tid + k * 256;
      const float q = cq[i];
      const float d = q - zp[i];
      op[i] = q;                                // z_q_st == z_q numerically
      lsum += d * d;
    }
  }
  lred[tid] = lsum;
  __syncthreads();
  for (int off = 128; off > 0; off >>= 1) {
    if (tid < off) lred[tid] += lred[tid + off];
    __syncthreads();
  }
  if (tid == 0) partials[blockIdx.x] = lred[0];
}

// deterministic loss finalize: loss = (1 + beta) * sum / (32*1024*512)
__global__ void __launch_bounds__(256)
vq_loss_finalize_kernel(const float* __restrict__ partials,
                        float* __restrict__ out_loss) {
  __shared__ float red[256];
  const int tid = threadIdx.x;
  red[tid] = partials[tid];
  __syncthreads();
  for (int off = 128; off > 0; off >>= 1) {
    if (tid < off) red[tid] += red[tid + off];
    __syncthreads();
  }
  if (tid == 0) out_loss[0] = 1.25f * red[0] / 16777216.0f;
}

// ---------------------------------------------------------------------------
extern "C" void kernel_launch(void* const* d_in, const int* in_sizes, int n_in,
                              void* d_out, int out_size, void* d_ws, size_t ws_size,
                              hipStream_t stream) {
  const float* z    = (const float*)d_in[0];   // [32,1024,512] f32
  const float* code = (const float*)d_in[1];   // [8192,512]    f32

  char* ws = (char*)d_ws;
  float*          c_norm   = (float*)(ws);                                   // 16 MB
  unsigned short* code_bf  = (unsigned short*)(ws + (size_t)16777216);       //  8 MB
  unsigned short* z_bf     = (unsigned short*)(ws + (size_t)16777216 + 8388608);      // 32 MB
  float*          partials = (float*)(ws + (size_t)16777216 + 8388608 + 33554432);    //  1 KB

  float* out      = (float*)d_out;
  float* out_zq   = out;                 // 16777216 floats
  float* out_loss = out + 16777216;      // 1 float
  float* out_idx  = out + 16777217;      // 32768 floats (indices as f32)

  hipLaunchKernelGGL(vq_normalize_kernel, dim3(NE), dim3(256), 0, stream,
                     code, c_norm, code_bf);
  hipLaunchKernelGGL(vq_normalize_kernel, dim3(ROWS), dim3(256), 0, stream,
                     z, (float*)nullptr, z_bf);
  hipLaunchKernelGGL(vq_argmax_gemm_kernel, dim3(ROWS / M_TILE), dim3(256), 0, stream,
                     z_bf, code_bf, c_norm, z, out_zq, out_idx, partials);
  hipLaunchKernelGGL(vq_loss_finalize_kernel, dim3(1), dim3(256), 0, stream,
                     partials, out_loss);
}
